// PhotoVoModel_13219909337346
// MI455X (gfx1250) — matrix operand order repaired and checked by reading, not compiled, not run on hardware
//
#include <hip/hip_runtime.h>

#define BN 8
#define HH 384
#define WW 384
#define NN 512
#define PP 8
#define PDIM 192
#define DDIM 128

typedef __bf16 bf16_t;
typedef bf16_t v16bf __attribute__((ext_vector_type(16)));
typedef short  v16s  __attribute__((ext_vector_type(16)));
typedef float  v8f   __attribute__((ext_vector_type(8)));
typedef unsigned int v4u __attribute__((ext_vector_type(4)));
typedef int    v4i   __attribute__((ext_vector_type(4)));
typedef int    v8i   __attribute__((ext_vector_type(8)));

#if defined(__HIP_DEVICE_COMPILE__) && __has_builtin(__builtin_amdgcn_tensor_load_to_lds)
#define USE_TDM 1
#else
#define USE_TDM 0
#endif

__device__ __forceinline__ unsigned short f2bf(float f) {
    unsigned int u = __float_as_uint(f);
    unsigned int r = u + 0x7FFFu + ((u >> 16) & 1u);   // round-to-nearest-even
    return (unsigned short)(r >> 16);
}
__device__ __forceinline__ float bf2f(unsigned short h) {
    return __uint_as_float(((unsigned int)h) << 16);
}

// ---------------------------------------------------------------------------
// Generic implicit-GEMM conv (NHWC bf16), fused scale/bias(+residual)(+relu).
// Block = 256 threads = 8 waves; block tile = 128(M) x 64(N); k-step = 32.
// Waves arranged 4(M) x 2(N); each wave owns a 32x32 subtile -> 4 WMMAs per
// k-step with full A/B fragment reuse (v_wmma_f32_16x16x32_bf16).
// B (weight) tile is DMA'd into LDS by the Tensor Data Mover (wave 0 issues
// tensor_load_to_lds, waits on TENSORcnt, then block barrier); A (im2col) tile
// is staged cooperatively with zero-padding for OOB / K-tail.
// Requirements: M % 128 == 0, Cout % 64 == 0 (true for every launch here).
// ---------------------------------------------------------------------------
__global__ __launch_bounds__(256)
void conv_wmma_kernel(const unsigned short* __restrict__ in,
                      const unsigned short* __restrict__ wgt,   // [K, Cout] bf16
                      const float* __restrict__ sc, const float* __restrict__ bi,
                      const unsigned short* __restrict__ res,   // NHWC bf16 or null
                      unsigned short* __restrict__ out,
                      int Bn, int Hin, int Win, int Cin,
                      int Hout, int Wout, int Cout,
                      int kh, int kw, int stride, int pad, int relu)
{
    __shared__ unsigned short As[128][34];
    __shared__ unsigned short Bs[32][64];    // unpadded: TDM writes packed tile

    const int tid  = threadIdx.x;
    const int lane = tid & 31;
    const int wv   = tid >> 5;
    const int m0   = blockIdx.x * 128;
    const int n0   = blockIdx.y * 64;
    const int mi   = (wv >> 1) * 32;         // 0,32,64,96
    const int ni   = (wv & 1) * 32;          // 0,32
    const int K    = kh * kw * Cin;
    const int hi8  = lane >> 4;              // 0 or 1
    const int l15  = lane & 15;

    v8f acc[2][2] = {};

    for (int k0 = 0; k0 < K; k0 += 32) {
        if (k0 + 32 < K)                     // prefetch next weight tile
            __builtin_prefetch(&wgt[(size_t)(k0 + 32) * Cout + n0], 0, 1);

#if USE_TDM
        // ---- B tile via Tensor Data Mover: 32(k) x 64(n), stride Cout ----
        if (wv == 0) {
            unsigned long long ga =
                (unsigned long long)(const void*)&wgt[(size_t)k0 * Cout + n0];
            unsigned int ldsa = (unsigned int)(unsigned long long)&Bs[0][0];
            int kRem = K - k0;               // rows beyond -> TDM OOB reads 0
            v4u g0;
            g0[0] = 1u;                                      // count=1 (user D#)
            g0[1] = ldsa;                                    // lds_addr
            g0[2] = (unsigned int)(ga & 0xFFFFFFFFu);        // global_addr lo
            g0[3] = (unsigned int)((ga >> 32) & 0x01FFFFFFu) // global_addr hi
                    | (2u << 30);                            // type = 2 (image)
            v8i g1;
            g1[0] = 0x00010000;                              // data_size=1 (2B)
            g1[1] = (int)(((unsigned)Cout & 0xFFFFu) << 16); // tensor_dim0 lo16
            g1[2] = (int)((((unsigned)Cout >> 16) & 0xFFFFu)
                    | (((unsigned)kRem & 0xFFFFu) << 16));   // dim0 hi / dim1 lo
            g1[3] = (int)((((unsigned)kRem >> 16) & 0xFFFFu)
                    | (64u << 16));                          // dim1 hi / tile_dim0=64
            g1[4] = 32;                                      // tile_dim1=32, tile_dim2=0
            g1[5] = Cout;                                    // tensor_dim0_stride lo
            g1[6] = 0;
            g1[7] = 0;
            v4i gz = {0, 0, 0, 0};
#if defined(__clang_major__) && __clang_major__ >= 23
            v8i gz8 = {0, 0, 0, 0, 0, 0, 0, 0};
            __builtin_amdgcn_tensor_load_to_lds(g0, g1, gz, gz, gz8, 0);
#else
            __builtin_amdgcn_tensor_load_to_lds(g0, g1, gz, gz, 0);
#endif
        }
#else
        // ---- fallback: cooperative B tile load ----
        for (int it = 0; it < 8; ++it) {     // 32*64 / 256
            int t  = tid + it * 256;
            int kk = k0 + (t >> 6);
            int n  = n0 + (t & 63);
            Bs[t >> 6][t & 63] = (kk < K) ? wgt[(size_t)kk * Cout + n]
                                          : (unsigned short)0;
        }
#endif

        // ---- stage A: 128 x 32 (im2col gather, zero-pad OOB / k>=K) ----
        for (int it = 0; it < 16; ++it) {    // 128*32 / 256
            int t  = tid + it * 256;
            int r  = t >> 5;
            int kk = k0 + (t & 31);
            unsigned short v = 0;
            if (kk < K) {
                int cell = kk / Cin;
                int ci   = kk - cell * Cin;
                int ky   = cell / kw;
                int kx   = cell - ky * kw;
                int m    = m0 + r;
                int wo   = m % Wout;
                int t2   = m / Wout;
                int ho   = t2 % Hout;
                int b    = t2 / Hout;
                int hi_  = ho * stride - pad + ky;
                int wi_  = wo * stride - pad + kx;
                if ((unsigned)hi_ < (unsigned)Hin && (unsigned)wi_ < (unsigned)Win)
                    v = in[((size_t)(b * Hin + hi_) * Win + wi_) * Cin + ci];
            }
            As[r][t & 31] = v;
        }

#if USE_TDM
        if (wv == 0) __builtin_amdgcn_s_wait_tensorcnt(0);
#endif
        __syncthreads();

        // ---- fragments (ISA VGPR layouts) + 4 WMMAs with A/B reuse ----
        v16s av[2], bv[2];
#pragma unroll
        for (int e = 0; e < 16; ++e) {
            int ka = (e & 7) + ((e >> 3) << 4) + (hi8 << 3);  // A 16x32 layout
            int kb = (e & 7) + ((e >> 3) << 4) + (hi8 << 3);  // B 32x16 layout
            av[0][e] = (short)As[mi + l15][ka];
            av[1][e] = (short)As[mi + 16 + l15][ka];
            bv[0][e] = (short)Bs[kb][ni + l15];
            bv[1][e] = (short)Bs[kb][ni + 16 + l15];
        }
#pragma unroll
        for (int i = 0; i < 2; ++i)
#pragma unroll
            for (int j = 0; j < 2; ++j)
                acc[i][j] = __builtin_amdgcn_wmma_f32_16x16x32_bf16(
                                false, __builtin_bit_cast(v16bf, av[i]),
                                false, __builtin_bit_cast(v16bf, bv[j]),
                                (short)0, acc[i][j], false, false);
        __syncthreads();
    }

    // ---- epilogue: BN scale/bias (+residual)(+relu), store bf16 ----
#pragma unroll
    for (int j = 0; j < 2; ++j) {
        int n = n0 + ni + j * 16 + l15;
        float s  = sc ? sc[n] : 1.0f;
        float bb = bi ? bi[n] : 0.0f;
#pragma unroll
        for (int i = 0; i < 2; ++i) {
#pragma unroll
            for (int r2 = 0; r2 < 8; ++r2) {
                int m = m0 + mi + i * 16 + r2 + (hi8 << 3);
                float v = acc[i][j][r2] * s + bb;
                if (res) v += bf2f(res[(size_t)m * Cout + n]);
                if (relu) v = fmaxf(v, 0.0f);
                out[(size_t)m * Cout + n] = f2bf(v);
            }
        }
    }
}

// ---------------------------------------------------------------------------
// Elementwise / small kernels
// ---------------------------------------------------------------------------
__global__ void prep_images_kernel(const float* __restrict__ im0,
                                   const float* __restrict__ im1,
                                   unsigned short* __restrict__ out6)
{
    const float mean[3] = {0.485f, 0.456f, 0.406f};
    const float istd[3] = {1.0f / 0.229f, 1.0f / 0.224f, 1.0f / 0.225f};
    int idx = blockIdx.x * blockDim.x + threadIdx.x;
    int total = BN * HH * WW * 6;
    if (idx >= total) return;
    int c6 = idx % 6;
    int x  = (idx / 6) % WW;
    int y  = (idx / (6 * WW)) % HH;
    int b  = idx / (6 * WW * HH);
    int c  = (c6 < 3) ? c6 : (c6 - 3);
    const float* src = (c6 < 3) ? im0 : im1;
    float v = src[(((size_t)b * 3 + c) * HH + y) * WW + x];
    out6[idx] = f2bf((v - mean[c]) * istd[c]);
}

__global__ void maxpool_kernel(const unsigned short* __restrict__ in,
                               unsigned short* __restrict__ out,
                               int Bn, int Hin, int Win, int C, int Hout, int Wout)
{
    int idx = blockIdx.x * blockDim.x + threadIdx.x;
    int total = Bn * Hout * Wout * C;
    if (idx >= total) return;
    int c = idx % C;
    int x = (idx / C) % Wout;
    int y = (idx / (C * Wout)) % Hout;
    int b = idx / (C * Wout * Hout);
    float mx = -3.4e38f;
    for (int dy = 0; dy < 3; ++dy)
        for (int dx = 0; dx < 3; ++dx) {
            int hi = y * 2 - 1 + dy, wi = x * 2 - 1 + dx;
            if ((unsigned)hi < (unsigned)Hin && (unsigned)wi < (unsigned)Win)
                mx = fmaxf(mx, bf2f(in[((size_t)(b * Hin + hi) * Win + wi) * C + c]));
        }
    out[idx] = f2bf(mx);
}

__global__ void wreorder_kernel(const float* __restrict__ w,
                                unsigned short* __restrict__ out,
                                int O, int I, int kh, int kw)
{
    int idx = blockIdx.x * blockDim.x + threadIdx.x;
    int K = kh * kw * I;
    if (idx >= K * O) return;
    int k = idx / O;
    int o = idx - k * O;
    int cell = k / I;
    int ci = k - cell * I;
    int ky = cell / kw;
    int kx = cell - ky * kw;
    out[idx] = f2bf(w[(((size_t)o * I + ci) * kh + ky) * kw + kx]);
}

__global__ void f32_to_bf16_kernel(const float* __restrict__ src,
                                   unsigned short* __restrict__ dst, int n)
{
    int idx = blockIdx.x * blockDim.x + threadIdx.x;
    if (idx < n) dst[idx] = f2bf(src[idx]);
}

// stable compaction of valid matches to the front (per batch), with gathers
__global__ void compact_kernel(const int* __restrict__ matches,
                               const float* __restrict__ kpts0,
                               const float* __restrict__ kpts1,
                               const float* __restrict__ scores0,
                               const float* __restrict__ scores1,
                               int* __restrict__ vso,
                               float* __restrict__ k0o, float* __restrict__ k1o,
                               float* __restrict__ scat)
{
    int b = blockIdx.x;
    if (threadIdx.x != 0) return;
    int cnt = 0;
    for (int pass = 0; pass < 2; ++pass)
        for (int n2 = 0; n2 < NN; ++n2) {
            int mm1 = matches[((size_t)b * NN + n2) * 2 + 1];
            int v = (mm1 > -1) ? 1 : 0;
            if ((pass == 0) == (v == 1)) {
                int j = cnt++;
                int mm0 = matches[((size_t)b * NN + n2) * 2 + 0];
                int m1c = v ? mm1 : 0;
                vso[b * NN + j] = v;
                k0o[((size_t)b * NN + j) * 2 + 0] = kpts0[((size_t)b * NN + mm0) * 2 + 0];
                k0o[((size_t)b * NN + j) * 2 + 1] = kpts0[((size_t)b * NN + mm0) * 2 + 1];
                k1o[((size_t)b * NN + j) * 2 + 0] = kpts1[((size_t)b * NN + m1c) * 2 + 0];
                k1o[((size_t)b * NN + j) * 2 + 1] = kpts1[((size_t)b * NN + m1c) * 2 + 1];
                scat[(size_t)b * 2 * NN + j]      = v ? scores0[b * NN + mm0] : -1.0f;
                scat[(size_t)b * 2 * NN + NN + j] = v ? scores1[b * NN + m1c] : -1.0f;
            }
        }
}

// patch extraction + LayerNorm(192), one block per (b, slot)
__global__ __launch_bounds__(PDIM)
void patch_ln1_kernel(const float* __restrict__ im0, const float* __restrict__ im1,
                      const float* __restrict__ k0o, const float* __restrict__ k1o,
                      const int* __restrict__ vso,
                      const float* __restrict__ g, const float* __restrict__ bta,
                      unsigned short* __restrict__ xout)
{
    __shared__ float sv[PDIM];
    __shared__ float stats[2];
    int row = blockIdx.x;
    int b = row / (2 * NN);
    int j = row % (2 * NN);
    int side = (j >= NN) ? 1 : 0;
    int jj = side ? (j - NN) : j;
    int e = threadIdx.x;
    int c  = e >> 6;
    int py = (e >> 3) & 7;
    int px = e & 7;
    float val = -1.0f;
    if (vso[b * NN + jj]) {
        const float* kp  = side ? &k1o[((size_t)b * NN + jj) * 2]
                                : &k0o[((size_t)b * NN + jj) * 2];
        const float* img = side ? im1 : im0;
        int cx = (int)rintf(kp[0]);
        int cy = (int)rintf(kp[1]);
        int xs = min(max(cx + px - 4, 0), WW - 1);
        int ys = min(max(cy + py - 4, 0), HH - 1);
        val = img[(((size_t)b * 3 + c) * HH + ys) * WW + xs];
    }
    sv[e] = val;
    __syncthreads();
    if (e == 0) {
        float s = 0.0f, s2 = 0.0f;
        for (int i = 0; i < PDIM; ++i) { s += sv[i]; s2 += sv[i] * sv[i]; }
        float mu = s / PDIM;
        stats[0] = mu;
        stats[1] = rsqrtf(fmaxf(s2 / PDIM - mu * mu, 0.0f) + 1e-5f);
    }
    __syncthreads();
    xout[(size_t)row * PDIM + e] = f2bf((val - stats[0]) * stats[1] * g[e] + bta[e]);
}

// LayerNorm(128) + score concat -> d_out [B,2N,129] fp32
__global__ __launch_bounds__(DDIM)
void ln2_out_kernel(const unsigned short* __restrict__ y,
                    const float* __restrict__ g, const float* __restrict__ bta,
                    const float* __restrict__ scat, float* __restrict__ out)
{
    __shared__ float sv[DDIM];
    __shared__ float stats[2];
    int row = blockIdx.x;
    int e = threadIdx.x;
    float v = bf2f(y[(size_t)row * DDIM + e]);
    sv[e] = v;
    __syncthreads();
    if (e == 0) {
        float s = 0.0f, s2 = 0.0f;
        for (int i = 0; i < DDIM; ++i) { s += sv[i]; s2 += sv[i] * sv[i]; }
        float mu = s / DDIM;
        stats[0] = mu;
        stats[1] = rsqrtf(fmaxf(s2 / DDIM - mu * mu, 0.0f) + 1e-5f);
    }
    __syncthreads();
    out[(size_t)row * (DDIM + 1) + e] = (v - stats[0]) * stats[1] * g[e] + bta[e];
    if (e == 0) out[(size_t)row * (DDIM + 1) + DDIM] = scat[row];
}

// layer4 NHWC bf16 [8,12,12,512] -> NCHW fp32
__global__ void nchw_out_kernel(const unsigned short* __restrict__ act,
                                float* __restrict__ out)
{
    int idx = blockIdx.x * blockDim.x + threadIdx.x;
    int total = BN * 512 * 12 * 12;
    if (idx >= total) return;
    int x = idx % 12;
    int y = (idx / 12) % 12;
    int c = (idx / 144) % 512;
    int b = idx / (144 * 512);
    out[idx] = bf2f(act[(((size_t)b * 12 + y) * 12 + x) * 512 + c]);
}

// ---------------------------------------------------------------------------
extern "C" void kernel_launch(void* const* d_in, const int* in_sizes, int n_in,
                              void* d_out, int out_size, void* d_ws, size_t ws_size,
                              hipStream_t stream)
{
    (void)in_sizes; (void)n_in; (void)out_size; (void)ws_size;

    const float* im0     = (const float*)d_in[0];
    const float* im1     = (const float*)d_in[1];
    const float* kpts0   = (const float*)d_in[2];
    const float* kpts1   = (const float*)d_in[3];
    const int*   matches = (const int*)d_in[4];
    const float* scores0 = (const float*)d_in[5];
    const float* scores1 = (const float*)d_in[6];
    const float* ln1g = (const float*)d_in[7];
    const float* ln1b = (const float*)d_in[8];
    const float* pw   = (const float*)d_in[9];
    const float* pb   = (const float*)d_in[10];
    const float* ln2g = (const float*)d_in[11];
    const float* ln2b = (const float*)d_in[12];

    char* ws = (char*)d_ws;
    size_t off = 0;
    auto alloc = [&](size_t bytes) -> void* {
        off = (off + 255) & ~(size_t)255;
        void* p = ws + off;
        off += bytes;
        return p;
    };

    unsigned short* in6  = (unsigned short*)alloc((size_t)BN * HH * WW * 6 * 2);
    unsigned short* actA = (unsigned short*)alloc((size_t)BN * 192 * 192 * 64 * 2);
    unsigned short* actB = (unsigned short*)alloc((size_t)BN * 192 * 192 * 64 * 2);
    unsigned short* actC = (unsigned short*)alloc((size_t)BN * 96 * 96 * 64 * 2);
    unsigned short* xln  = (unsigned short*)alloc((size_t)BN * 2 * NN * PDIM * 2);
    unsigned short* ygem = (unsigned short*)alloc((size_t)BN * 2 * NN * DDIM * 2);
    float* scat = (float*)alloc((size_t)BN * 2 * NN * 4);
    int*   vso  = (int*)alloc((size_t)BN * NN * 4);
    float* k0o  = (float*)alloc((size_t)BN * NN * 2 * 4);
    float* k1o  = (float*)alloc((size_t)BN * NN * 2 * 4);

    auto wconv = [&](const float* w, int O, int I, int k) -> unsigned short* {
        int K = k * k * I;
        unsigned short* dst = (unsigned short*)alloc((size_t)K * O * 2);
        int total = K * O;
        wreorder_kernel<<<(total + 255) / 256, 256, 0, stream>>>(w, dst, O, I, k, k);
        return dst;
    };

    // ---- parse enc_params (insertion order) & convert weights ----
    int ip = 13;
    const float* c1w  = (const float*)d_in[ip++];
    const float* bn1s = (const float*)d_in[ip++];
    const float* bn1b = (const float*)d_in[ip++];
    unsigned short* Wc1 = wconv(c1w, 64, 6, 7);

    struct BlkW {
        unsigned short *c1, *c2, *dc;
        const float *b1s, *b1b, *b2s, *b2b, *dbs, *dbb;
        int has_ds;
    } blk[8];
    const int chans[5] = {64, 64, 128, 256, 512};
    int bix = 0;
    for (int li = 0; li < 4; ++li)
        for (int bbi = 0; bbi < 2; ++bbi, ++bix) {
            int cin  = (bbi == 0) ? chans[li] : chans[li + 1];
            int cout = chans[li + 1];
            BlkW& Bk = blk[bix];
            const float* w1 = (const float*)d_in[ip++];
            Bk.b1s = (const float*)d_in[ip++];
            Bk.b1b = (const float*)d_in[ip++];
            const float* w2 = (const float*)d_in[ip++];
            Bk.b2s = (const float*)d_in[ip++];
            Bk.b2b = (const float*)d_in[ip++];
            Bk.c1 = wconv(w1, cout, cin, 3);
            Bk.c2 = wconv(w2, cout, cout, 3);
            Bk.has_ds = (bbi == 0 && li > 0);
            if (Bk.has_ds) {
                const float* wd = (const float*)d_in[ip++];
                Bk.dbs = (const float*)d_in[ip++];
                Bk.dbb = (const float*)d_in[ip++];
                Bk.dc = wconv(wd, cout, cin, 1);
            } else {
                Bk.dc = nullptr; Bk.dbs = nullptr; Bk.dbb = nullptr;
            }
        }
    unsigned short* Wp = (unsigned short*)alloc((size_t)PDIM * DDIM * 2);
    f32_to_bf16_kernel<<<(PDIM * DDIM + 255) / 256, 256, 0, stream>>>(pw, Wp, PDIM * DDIM);

    auto conv = [&](const unsigned short* i_, const unsigned short* w_,
                    const float* s_, const float* b_, const unsigned short* r_,
                    unsigned short* o_, int Bn, int Hin, int Win, int Cin,
                    int Hout, int Wout, int Cout, int k, int stride, int pad, int relu) {
        int M = Bn * Hout * Wout;
        dim3 g(M / 128, Cout / 64);
        conv_wmma_kernel<<<g, 256, 0, stream>>>(i_, w_, s_, b_, r_, o_,
            Bn, Hin, Win, Cin, Hout, Wout, Cout, k, k, stride, pad, relu);
    };

    // ---- image trunk ----
    {
        int total = BN * HH * WW * 6;
        prep_images_kernel<<<(total + 255) / 256, 256, 0, stream>>>(im0, im1, in6);
    }
    conv(in6, Wc1, bn1s, bn1b, nullptr, actA, BN, 384, 384, 6, 192, 192, 64, 7, 2, 3, 1);
    {
        int total = BN * 96 * 96 * 64;
        maxpool_kernel<<<(total + 255) / 256, 256, 0, stream>>>(actA, actB,
                                                                BN, 192, 192, 64, 96, 96);
    }

    int Hc = 96, Cc = 64;   // current tensor lives in actB
    bix = 0;
    for (int li = 0; li < 4; ++li)
        for (int bbi = 0; bbi < 2; ++bbi, ++bix) {
            BlkW& Bk = blk[bix];
            int cout = chans[li + 1];
            int stride = (li > 0 && bbi == 0) ? 2 : 1;
            int Ho = Hc / stride;
            const unsigned short* skip;
            if (Bk.has_ds) {
                conv(actB, Bk.dc, Bk.dbs, Bk.dbb, nullptr, actC,
                     BN, Hc, Hc, Cc, Ho, Ho, cout, 1, stride, 0, 0);
                skip = actC;
            } else {
                skip = actB;   // identity: out==res aliasing is per-element safe
            }
            conv(actB, Bk.c1, Bk.b1s, Bk.b1b, nullptr, actA,
                 BN, Hc, Hc, Cc, Ho, Ho, cout, 3, stride, 1, 1);
            conv(actA, Bk.c2, Bk.b2s, Bk.b2b, skip, actB,
                 BN, Ho, Ho, cout, Ho, Ho, cout, 3, 1, 1, 1);
            Hc = Ho; Cc = cout;
        }

    float* out = (float*)d_out;
    const size_t patch_elems = (size_t)BN * 2 * NN * (DDIM + 1);
    {
        int total = BN * 512 * 12 * 12;
        nchw_out_kernel<<<(total + 255) / 256, 256, 0, stream>>>(actB, out + patch_elems);
    }

    // ---- patch branch ----
    compact_kernel<<<BN, 1, 0, stream>>>(matches, kpts0, kpts1, scores0, scores1,
                                         vso, k0o, k1o, scat);
    patch_ln1_kernel<<<BN * 2 * NN, PDIM, 0, stream>>>(im0, im1, k0o, k1o, vso,
                                                       ln1g, ln1b, xln);
    // PatchEncoder linear as 1x1 "conv": M=8192, K=192, N=128
    conv(xln, Wp, nullptr, pb, nullptr, ygem, BN * 2 * NN, 1, 1, PDIM, 1, 1, DDIM,
         1, 1, 0, 0);
    ln2_out_kernel<<<BN * 2 * NN, DDIM, 0, stream>>>(ygem, ln2g, ln2b, scat, out);
}
